// Pair_24739011625868
// MI455X (gfx1250) — matrix operand order repaired and checked by reading, not compiled, not run on hardware
//
#include <hip/hip_runtime.h>
#include <stdint.h>

typedef __bf16 bf16_t;
typedef __attribute__((ext_vector_type(16))) __bf16 v16bf;
typedef __attribute__((ext_vector_type(4)))  __bf16 v4bf;
typedef __attribute__((ext_vector_type(8)))  float  v8f;
typedef __attribute__((ext_vector_type(4)))  unsigned int uint4v;
typedef __attribute__((ext_vector_type(8)))  int int8v;
typedef __attribute__((ext_vector_type(4)))  int int4v;

#define DIM_IN 2048
#define LATENT 1024
#define N_ROIS 4096
#define N_BANK 8192
#define LN_EPS 1e-5f

#if defined(__has_builtin)
#  if __has_builtin(__builtin_amdgcn_tensor_load_to_lds)
#    define HAVE_TDM 1
#  else
#    define HAVE_TDM 0
#  endif
#else
#  define HAVE_TDM 0
#endif

union Frag16 { v16bf v; uint4v u[2]; };

__device__ __forceinline__ uint4v ld16g(const bf16_t* p) {
  return *reinterpret_cast<const uint4v*>(p);
}

#if HAVE_TDM
// Issue one TDM descriptor: load a [64 rows x 64 cols] bf16 tile (row stride
// ldw elements) from global into contiguous LDS at lds_addr. Tile dims ==
// tensor dims so no OOB clipping occurs. D# layout per CDNA5 ISA ch.8.
__device__ __forceinline__ void tdm_load_tile(const void* gsrc, unsigned lds_addr,
                                              unsigned ldw_elems) {
  const unsigned long long ga = (unsigned long long)reinterpret_cast<uintptr_t>(gsrc);
  uint4v g0;
  g0[0] = 1u;                                                // count=1, user D#
  g0[1] = lds_addr;                                          // LDS byte address
  g0[2] = (unsigned)ga;                                      // global_addr[31:0]
  g0[3] = (unsigned)((ga >> 32) & 0x01FFFFFFull) | (2u << 30); // ga[56:32] | type=2
  int8v g1;
  g1[0] = (int)(1u << 16);      // workgroup_mask=0, data_size=1 (2 bytes)
  g1[1] = (int)(64u << 16);     // tensor_dim0 = 64
  g1[2] = (int)(64u << 16);     // tensor_dim0_hi=0 | tensor_dim1 = 64
  g1[3] = (int)(64u << 16);     // tensor_dim1_hi=0 | tile_dim0 = 64
  g1[4] = 64;                   // tile_dim1 = 64, tile_dim2 = 0
  g1[5] = (int)ldw_elems;       // tensor_dim0_stride (elements)
  g1[6] = 0;
  g1[7] = 0;
  int4v gz = {};
#if __clang_major__ >= 23
  int8v gz8 = {};
  __builtin_amdgcn_tensor_load_to_lds(g0, g1, gz, gz, gz8, 0);
#else
  __builtin_amdgcn_tensor_load_to_lds(g0, g1, gz, gz, 0);
#endif
}
#endif

// ---------------- f32 -> bf16 conversion (4 elems / thread) ----------------
__global__ void __launch_bounds__(256) cvt_bf16_k(const float* __restrict__ s,
                                                  bf16_t* __restrict__ d, long n) {
  long i = ((long)blockIdx.x * 256 + threadIdx.x) * 4;
  if (i + 3 < n) {
    const float4 v = *reinterpret_cast<const float4*>(s + i);
    v4bf o = { (bf16_t)v.x, (bf16_t)v.y, (bf16_t)v.z, (bf16_t)v.w };
    *reinterpret_cast<v4bf*>(d + i) = o;
  }
}

// ---------------- bf16 WMMA GEMM: Out = A[M,K] * W[N,K]^T * scale + bias ----
// OUT_MODE: 0 = f32 [M,N], 1 = bf16 [M,N], 2 = bf16 transposed [N,M]
// 256 threads = 8 waves. Wave w: rows [m0+64w, m0+64w+64), cols [n0, n0+64):
// 4 A-frags x 4 B-frags = 16 WMMA per 32-wide K-chunk; each LDS B-fragment
// feeds 4 WMMAs. K-step 64 -> 32 WMMAs per (2 barriers + 1 TDM issue).
// W tile (64x64 bf16, 8KB) double-buffered in LDS; filled by the Tensor Data
// Mover (async DMA overlapped with WMMAs on the other buffer) when available.
// Requires M%512==0, N%64==0, K%64==0 (true for all calls here).
template<int OUT_MODE>
__global__ void __launch_bounds__(256) gemm_bf16_k(
    const bf16_t* __restrict__ A, int lda,
    const bf16_t* __restrict__ W, int ldw,
    const float*  __restrict__ bias,
    void* __restrict__ Out, int M, int N, int K, float scale)
{
  __shared__ __align__(16) bf16_t wtile[2][64][64];   // 2 x 8KB
  const int tid  = threadIdx.x;
  const int wave = tid >> 5;
  const int lane = tid & 31;
  const int l16  = lane & 15;
  const int hi   = lane >> 4;            // half-wave K-split per ISA layout
  const int n0   = blockIdx.x * 64;
  const int m0   = blockIdx.y * 512 + wave * 64;

  v8f acc[4][4] = {};

  const bf16_t* arow[4];
  #pragma unroll
  for (int a = 0; a < 4; ++a)
    arow[a] = A + (size_t)(m0 + a * 16 + l16) * lda;
  const bf16_t* wbase = W + (size_t)n0 * ldw;

#if HAVE_TDM
  const unsigned ldsbase = (unsigned)(uintptr_t)(&wtile[0][0][0]);
  if (wave == 0) tdm_load_tile(wbase, ldsbase, (unsigned)ldw);   // prologue: tile 0
#else
  // fallback: 512 x 16B chunks, 2 per thread
  const int r0 = tid >> 3, c0 = (tid & 7) * 8;            // chunk tid
  const int r1 = (tid + 256) >> 3, c1 = ((tid + 256) & 7) * 8;
  const bf16_t* wsrc0 = wbase + (size_t)r0 * ldw + c0;
  const bf16_t* wsrc1 = wbase + (size_t)r1 * ldw + c1;
#endif

  int cur = 0;
  for (int k0 = 0; k0 < K; k0 += 64, cur ^= 1) {
    __syncthreads();                     // everyone done reading buf[cur^1]
#if HAVE_TDM
    if (wave == 0) {
      const int knext = (k0 + 64 < K) ? (k0 + 64) : 0;  // dummy re-issue on last iter
      tdm_load_tile(wbase + knext, ldsbase + (unsigned)((cur ^ 1) * 8192), (unsigned)ldw);
      __builtin_amdgcn_s_wait_tensorcnt(1);  // in-order: tile for buf[cur] complete
      asm volatile("" ::: "memory");
    }
#else
    *reinterpret_cast<uint4v*>(&wtile[cur][r0][c0]) = ld16g(wsrc0 + k0);
    *reinterpret_cast<uint4v*>(&wtile[cur][r1][c1]) = ld16g(wsrc1 + k0);
#endif
    __syncthreads();                     // buf[cur] visible to all waves

    if (k0 + 64 < K) {                   // speculative prefetch of next A tile
      #pragma unroll
      for (int a = 0; a < 4; ++a) __builtin_prefetch(arow[a] + k0 + 64, 0, 1);
    }

    #pragma unroll
    for (int kc = 0; kc < 64; kc += 32) {
      // A fragments (16x32): lanes 0-15 K 0..7 & 16..23; lanes 16-31 K 8..15 & 24..31
      Frag16 af[4];
      #pragma unroll
      for (int a = 0; a < 4; ++a) {
        af[a].u[0] = ld16g(arow[a] + k0 + kc + hi * 8);
        af[a].u[1] = ld16g(arow[a] + k0 + kc + 16 + hi * 8);
      }
      // B fragments (32x16): lane holds col n=l16, same K order as A (8h+0..7, 16+8h+0..7)
      #pragma unroll
      for (int nt = 0; nt < 4; ++nt) {
        Frag16 bf;
        const bf16_t* lp = &wtile[cur][nt * 16 + l16][kc + hi * 8];
        bf.u[0] = *(const uint4v*)lp;
        bf.u[1] = *(const uint4v*)(lp + 16);
        #pragma unroll
        for (int a = 0; a < 4; ++a)
          acc[a][nt] = __builtin_amdgcn_wmma_f32_16x16x32_bf16(
              false, af[a].v, false, bf.v, (short)0, acc[a][nt], false, false);
      }
    }
  }

  // Epilogue. C layout: lane l<16 -> col l, rows 0..7 ; lane l>=16 -> col l-16, rows 8..15
  #pragma unroll
  for (int a = 0; a < 4; ++a) {
    const int mbase = m0 + a * 16 + hi * 8;
    #pragma unroll
    for (int nt = 0; nt < 4; ++nt) {
      const v8f av = acc[a][nt];
      const int n = n0 + nt * 16 + l16;
      const float bb = bias ? bias[n] : 0.0f;
      if (OUT_MODE == 0) {
        float* o = (float*)Out + (size_t)mbase * N + n;
        #pragma unroll
        for (int r = 0; r < 8; ++r) o[(size_t)r * N] = av[r] * scale + bb;
      } else if (OUT_MODE == 1) {
        bf16_t* o = (bf16_t*)Out + (size_t)mbase * N + n;
        #pragma unroll
        for (int r = 0; r < 8; ++r) o[(size_t)r * N] = (bf16_t)(av[r] * scale + bb);
      } else {
        bf16_t* o = (bf16_t*)Out + (size_t)n * M + mbase; // contiguous 8 bf16 per lane
        #pragma unroll
        for (int r = 0; r < 8; ++r) o[r] = (bf16_t)(av[r] * scale + bb);
      }
    }
  }
}

// ---------------- block reductions (256 threads = 8 waves of 32) ------------
__device__ __forceinline__ float blk_red_max(float v, volatile float* sh) {
  #pragma unroll
  for (int off = 16; off; off >>= 1) v = fmaxf(v, __shfl_xor(v, off, 32));
  if ((threadIdx.x & 31) == 0) sh[threadIdx.x >> 5] = v;
  __syncthreads();
  float r = sh[0];
  #pragma unroll
  for (int i = 1; i < 8; ++i) r = fmaxf(r, sh[i]);
  __syncthreads();
  return r;
}
__device__ __forceinline__ float blk_red_sum(float v, volatile float* sh) {
  #pragma unroll
  for (int off = 16; off; off >>= 1) v += __shfl_xor(v, off, 32);
  if ((threadIdx.x & 31) == 0) sh[threadIdx.x >> 5] = v;
  __syncthreads();
  float r = 0.f;
  #pragma unroll
  for (int i = 0; i < 8; ++i) r += sh[i];
  __syncthreads();
  return r;
}

// ---------------- row softmax: f32 scores -> normalized bf16 probs ----------
__global__ void __launch_bounds__(256) softmax_k(const float* __restrict__ S,
                                                 bf16_t* __restrict__ P, int ncols) {
  __shared__ float sh[8];
  const size_t row = blockIdx.x;
  const float* s = S + row * (size_t)ncols;
  bf16_t* p = P + row * (size_t)ncols;
  float mx = -3.4e38f;
  for (int c = threadIdx.x; c < ncols; c += 256) mx = fmaxf(mx, s[c]);
  mx = blk_red_max(mx, sh);
  float sum = 0.f;
  for (int c = threadIdx.x; c < ncols; c += 256) sum += __expf(s[c] - mx);
  sum = blk_red_sum(sum, sh);
  const float inv = 1.0f / sum;
  for (int c = threadIdx.x; c < ncols; c += 256)
    p[c] = (bf16_t)(__expf(s[c] - mx) * inv);
}

// ---------------- cpair = f1*f2 -> LayerNorm -> PReLU -> bf16 ---------------
__global__ void __launch_bounds__(256) cpair_ln_k(const float* __restrict__ F1,
    const float* __restrict__ F2, const float* __restrict__ lnw,
    const float* __restrict__ lnb, const float* __restrict__ pa,
    bf16_t* __restrict__ X) {
  __shared__ float sh[8];
  const size_t row = blockIdx.x;
  const float* f1 = F1 + row * LATENT;
  const float* f2 = F2 + row * LATENT;
  bf16_t* x = X + row * LATENT;
  float v[4], sum = 0.f, sq = 0.f;
  #pragma unroll
  for (int i = 0; i < 4; ++i) {
    const int c = threadIdx.x + i * 256;
    v[i] = f1[c] * f2[c];
    sum += v[i]; sq += v[i] * v[i];
  }
  sum = blk_red_sum(sum, sh);
  sq  = blk_red_sum(sq, sh);
  const float mu   = sum * (1.0f / LATENT);
  const float var  = sq * (1.0f / LATENT) - mu * mu;
  const float rstd = rsqrtf(var + LN_EPS);
  const float a = pa[0];
  #pragma unroll
  for (int i = 0; i < 4; ++i) {
    const int c = threadIdx.x + i * 256;
    float y = (v[i] - mu) * rstd * lnw[c] + lnb[c];
    y = (y >= 0.f) ? y : a * y;
    x[c] = (bf16_t)y;
  }
}

extern "C" void kernel_launch(void* const* d_in, const int* in_sizes, int n_in,
                              void* d_out, int out_size, void* d_ws, size_t ws_size,
                              hipStream_t stream) {
  const float* feat  = (const float*)d_in[0];
  const float* kbank = (const float*)d_in[1];
  const float* Wc1 = (const float*)d_in[2];  const float* bc1 = (const float*)d_in[3];
  const float* Wc2 = (const float*)d_in[4];  const float* bc2 = (const float*)d_in[5];
  const float* Wc3 = (const float*)d_in[6];  const float* bc3 = (const float*)d_in[7];
  const float* Wd1 = (const float*)d_in[8];  const float* bd1 = (const float*)d_in[9];
  const float* Wd2 = (const float*)d_in[10]; const float* bd2 = (const float*)d_in[11];
  const float* Wd3 = (const float*)d_in[12]; const float* bd3 = (const float*)d_in[13];
  const float* lnw = (const float*)d_in[14]; const float* lnb = (const float*)d_in[15];
  const float* pa  = (const float*)d_in[16];
  const float* Wffn= (const float*)d_in[17]; const float* bffn= (const float*)d_in[18];

  // workspace carve-out (~390 MB total)
  char* w = (char*)d_ws;
  auto take = [&](size_t bytes) -> char* {
    char* p = w; w += (bytes + 255) & ~(size_t)255; return p;
  };
  bf16_t* featb = (bf16_t*)take((size_t)N_ROIS * DIM_IN * 2);
  bf16_t* kb    = (bf16_t*)take((size_t)N_BANK * DIM_IN * 2);
  bf16_t* Wc1b  = (bf16_t*)take((size_t)LATENT * DIM_IN * 2);
  bf16_t* Wc2b  = (bf16_t*)take((size_t)LATENT * DIM_IN * 2);
  bf16_t* Wc3b  = (bf16_t*)take((size_t)LATENT * DIM_IN * 2);
  bf16_t* Wd1b  = (bf16_t*)take((size_t)LATENT * DIM_IN * 2);
  bf16_t* Wd2b  = (bf16_t*)take((size_t)LATENT * DIM_IN * 2);
  bf16_t* Wd3b  = (bf16_t*)take((size_t)LATENT * DIM_IN * 2);
  bf16_t* Wffnb = (bf16_t*)take((size_t)DIM_IN * LATENT * 2);
  bf16_t* q1b   = (bf16_t*)take((size_t)N_ROIS * LATENT * 2);
  bf16_t* q2b   = (bf16_t*)take((size_t)N_ROIS * LATENT * 2);
  bf16_t* k1b   = (bf16_t*)take((size_t)N_BANK * LATENT * 2);
  bf16_t* k2b   = (bf16_t*)take((size_t)N_BANK * LATENT * 2);
  bf16_t* v1T   = (bf16_t*)take((size_t)LATENT * N_BANK * 2); // [L, M] = V^T
  bf16_t* v2T   = (bf16_t*)take((size_t)LATENT * N_BANK * 2);
  float*  scores= (float*) take((size_t)N_ROIS * N_BANK * 4); // reused per branch
  bf16_t* pbuf  = (bf16_t*)take((size_t)N_ROIS * N_BANK * 2); // reused per branch
  float*  f1    = (float*) take((size_t)N_ROIS * LATENT * 4);
  float*  f2    = (float*) take((size_t)N_ROIS * LATENT * 4);
  bf16_t* xln   = (bf16_t*)take((size_t)N_ROIS * LATENT * 2);

  auto cvt = [&](const float* s, bf16_t* d, long n) {
    cvt_bf16_k<<<(unsigned)(n / 1024), 256, 0, stream>>>(s, d, n);
  };
  cvt(feat,  featb, (long)N_ROIS * DIM_IN);
  cvt(kbank, kb,    (long)N_BANK * DIM_IN);
  cvt(Wc1, Wc1b, (long)LATENT * DIM_IN);  cvt(Wc2, Wc2b, (long)LATENT * DIM_IN);
  cvt(Wc3, Wc3b, (long)LATENT * DIM_IN);  cvt(Wd1, Wd1b, (long)LATENT * DIM_IN);
  cvt(Wd2, Wd2b, (long)LATENT * DIM_IN);  cvt(Wd3, Wd3b, (long)LATENT * DIM_IN);
  cvt(Wffn, Wffnb, (long)DIM_IN * LATENT);

  const dim3 blk(256);
  // ---- projections ----
  { dim3 g(LATENT / 64, N_ROIS / 512);
    gemm_bf16_k<1><<<g, blk, 0, stream>>>(featb, DIM_IN, Wc1b, DIM_IN, bc1, q1b, N_ROIS, LATENT, DIM_IN, 1.0f);
    gemm_bf16_k<1><<<g, blk, 0, stream>>>(featb, DIM_IN, Wd1b, DIM_IN, bd1, q2b, N_ROIS, LATENT, DIM_IN, 1.0f); }
  { dim3 g(LATENT / 64, N_BANK / 512);
    gemm_bf16_k<1><<<g, blk, 0, stream>>>(kb, DIM_IN, Wc2b, DIM_IN, bc2, k1b, N_BANK, LATENT, DIM_IN, 1.0f);
    gemm_bf16_k<1><<<g, blk, 0, stream>>>(kb, DIM_IN, Wd2b, DIM_IN, bd2, k2b, N_BANK, LATENT, DIM_IN, 1.0f);
    gemm_bf16_k<2><<<g, blk, 0, stream>>>(kb, DIM_IN, Wc3b, DIM_IN, bc3, v1T, N_BANK, LATENT, DIM_IN, 1.0f);
    gemm_bf16_k<2><<<g, blk, 0, stream>>>(kb, DIM_IN, Wd3b, DIM_IN, bd3, v2T, N_BANK, LATENT, DIM_IN, 1.0f); }

  const float sscale = 0.03125f; // 1/sqrt(LATENT)
  // ---- branch 1: S = q1 k1^T /32 ; p = softmax(S) ; f1 = p V1 ----
  { dim3 g(N_BANK / 64, N_ROIS / 512);
    gemm_bf16_k<0><<<g, blk, 0, stream>>>(q1b, LATENT, k1b, LATENT, nullptr, scores, N_ROIS, N_BANK, LATENT, sscale); }
  softmax_k<<<N_ROIS, blk, 0, stream>>>(scores, pbuf, N_BANK);
  { dim3 g(LATENT / 64, N_ROIS / 512);
    gemm_bf16_k<0><<<g, blk, 0, stream>>>(pbuf, N_BANK, v1T, N_BANK, nullptr, f1, N_ROIS, LATENT, N_BANK, 1.0f); }
  // ---- branch 2 ----
  { dim3 g(N_BANK / 64, N_ROIS / 512);
    gemm_bf16_k<0><<<g, blk, 0, stream>>>(q2b, LATENT, k2b, LATENT, nullptr, scores, N_ROIS, N_BANK, LATENT, sscale); }
  softmax_k<<<N_ROIS, blk, 0, stream>>>(scores, pbuf, N_BANK);
  { dim3 g(LATENT / 64, N_ROIS / 512);
    gemm_bf16_k<0><<<g, blk, 0, stream>>>(pbuf, N_BANK, v2T, N_BANK, nullptr, f2, N_ROIS, LATENT, N_BANK, 1.0f); }

  // ---- gate + LayerNorm + PReLU -> bf16 ----
  cpair_ln_k<<<N_ROIS, blk, 0, stream>>>(f1, f2, lnw, lnb, pa, xln);

  // ---- FFN: out = xln @ Wffn^T + bffn (f32 out) ----
  { dim3 g(DIM_IN / 64, N_ROIS / 512);
    gemm_bf16_k<0><<<g, blk, 0, stream>>>(xln, LATENT, Wffnb, LATENT, bffn, (float*)d_out, N_ROIS, DIM_IN, LATENT, 1.0f); }
}
